// CLD_SimSiam_50835232915742
// MI455X (gfx1250) — matrix operand order repaired and checked by reading, not compiled, not run on hardware
//
#include <hip/hip_runtime.h>

#define B_ROWS 8192
#define D_IN   2048
#define D_HID  512
#define D_OUT  2048
#define D_CLD  1024
#define KCLUST 10
#define NITERS 10
#define CLD_T  0.07f
#define BN_EPS 1e-5f

typedef __bf16 bf16_t;
typedef __attribute__((ext_vector_type(16))) __bf16 v16bf;
typedef __attribute__((ext_vector_type(8)))  __bf16 v8bf;
typedef __attribute__((ext_vector_type(8)))  float  v8f;

union FragAB { v16bf v; v8bf h[2]; };

// ---------------------------------------------------------------------------
// Generic zero
// ---------------------------------------------------------------------------
__global__ void zero_kernel(float* __restrict__ p, int n) {
    int i = blockIdx.x * blockDim.x + threadIdx.x;
    if (i < n) p[i] = 0.0f;
}

// ---------------------------------------------------------------------------
// Column inverse-norms of Wc (2048 x 1024), for l2norm(Wc, axis=0)
// ---------------------------------------------------------------------------
__global__ void colnorm_kernel(const float* __restrict__ Wc, float* __restrict__ cScale) {
    int col = blockIdx.x * blockDim.x + threadIdx.x;
    if (col >= D_CLD) return;
    float s = 0.0f;
    for (int r = 0; r < D_IN; ++r) {
        float v = Wc[(size_t)r * D_CLD + col];
        s += v * v;
    }
    cScale[col] = 1.0f / fmaxf(sqrtf(s), 1e-12f);
}

// ---------------------------------------------------------------------------
// Pack a row-major f32 K x N weight matrix into WMMA B-fragment layout (bf16).
// Fragment f = (k0/32)*nTiles + n0/16 holds 32 lanes x 16 bf16 contiguous.
// Lane L, element i maps to  n = n0 + (L&15),  k = k0 + ((L>=16)?16:0) + i.
// Optional per-column scale (for l2norm(Wc,0)).
// ---------------------------------------------------------------------------
__global__ void pack_b_kernel(const float* __restrict__ W, const float* __restrict__ colScale,
                              bf16_t* __restrict__ Bp, int K, int N) {
    int idx = blockIdx.x * blockDim.x + threadIdx.x;
    int nTiles = N >> 4;
    int frags = (K >> 5) * nTiles;
    if (idx >= frags * 32) return;
    int frag = idx >> 5, lane = idx & 31;
    int kt = frag / nTiles, nt = frag - kt * nTiles;
    int n  = (nt << 4) + (lane & 15);
    int kb = (kt << 5) + ((lane >> 4) << 4);
    float s = colScale ? colScale[n] : 1.0f;
    bf16_t* out = Bp + (size_t)frag * 512 + lane * 16;
#pragma unroll
    for (int i = 0; i < 16; ++i)
        out[i] = (bf16_t)(W[(size_t)(kb + i) * N + n] * s);
}

// ---------------------------------------------------------------------------
// Per-row prep for y: row L2 norm, emit bf16 copy + bf16 normalized copy.
// One wave per row (wave32).
// ---------------------------------------------------------------------------
__global__ __launch_bounds__(256)
void rowprep_kernel(const float* __restrict__ Y, bf16_t* __restrict__ Ybf,
                    bf16_t* __restrict__ Ynbf) {
    int lane = threadIdx.x & 31;
    int gw = (blockIdx.x * blockDim.x + threadIdx.x) >> 5;
    int nw = (gridDim.x * blockDim.x) >> 5;
    for (int row = gw; row < B_ROWS; row += nw) {
        const float* y = Y + (size_t)row * D_IN;
        float s = 0.0f;
        for (int i = lane; i < D_IN; i += 32) { float v = y[i]; s += v * v; }
#pragma unroll
        for (int m = 16; m >= 1; m >>= 1) s += __shfl_xor(s, m, 32);
        float inv = 1.0f / fmaxf(sqrtf(s), 1e-12f);
        bf16_t* ob = Ybf  + (size_t)row * D_IN;
        bf16_t* on = Ynbf + (size_t)row * D_IN;
        for (int i = lane; i < D_IN; i += 32) {
            float v = y[i];
            ob[i] = (bf16_t)v;
            on[i] = (bf16_t)(v * inv);
        }
    }
}

// ---------------------------------------------------------------------------
// WMMA GEMM with async-LDS B staging.
// Block = 8 waves = 128x64 output tile; wave w owns rows [16w, 16w+16) of the
// block's row group and the full 64-col strip (4 n-tiles -> 4 accumulators).
// Per k-step (K+=32) the block's 4 B-fragments (4 KB, contiguous in packed
// layout) are staged global->LDS with GLOBAL_LOAD_ASYNC_TO_LDS_B128 (16 B per
// thread), double buffered: wait own ASYNCcnt -> barrier -> issue next -> MMA.
// An empty asm fence keeps all 4 fragments live so all 8 ds_load_b128 issue
// before a single s_wait_dscnt, and the 4 WMMAs run back-to-back.
// LDS addresses are XOR-swizzled on bit8<->bit4 so the 32 B/lane fragment
// reads are bank-conflict-minimal (each half-wave covers all 64 banks once).
// ---------------------------------------------------------------------------
__global__ __launch_bounds__(256)
void gemm_bf16_wmma(const bf16_t* __restrict__ A, const bf16_t* __restrict__ Bp,
                    const float* __restrict__ bias, float* __restrict__ C,
                    int M, int N, int K) {
    __shared__ __attribute__((aligned(16))) bf16_t ldsB[2][4 * 512];

    const int tid  = threadIdx.x;
    const int lane = tid & 31;
    const int wave = tid >> 5;
    const int nTiles  = N >> 4;
    const int nGroups = N >> 6;                  // 64-col strips
    const int mGroup = blockIdx.x / nGroups;     // 8 m-tiles per block
    const int ng     = blockIdx.x - mGroup * nGroups;
    const int mTile  = mGroup * 8 + wave;

    const int m   = (mTile << 4) + (lane & 15);
    const int sel = (lane >> 4) << 4;            // 0 or 16
    const bf16_t* arow = A + (size_t)m * K;
    const bf16_t* bsrc = Bp + (size_t)(ng << 2) * 512;   // frags for this strip, k-tile 0
    const size_t  bstep = (size_t)nTiles * 512;          // elements per k-tile

    // async-copy dest (swizzled): thread t -> frag f=t/64, chunk w=t%64
    const int f = tid >> 6, w = tid & 63;
    const unsigned dstOff = (unsigned)(f * 1024 + ((w ^ ((w >> 4) & 1)) << 4));
    const unsigned ldsBase0 = (unsigned)(size_t)(&ldsB[0][0]);
    const unsigned ldsBase1 = (unsigned)(size_t)(&ldsB[1][0]);

    // prologue: stage k-tile 0 into buffer 0
    {
        unsigned long long g = (unsigned long long)(size_t)(bsrc + tid * 8);
        asm volatile("global_load_async_to_lds_b128 %0, %1, off"
                     :: "v"(ldsBase0 + dstOff), "v"(g) : "memory");
    }
    const bf16_t* bnext = bsrc + bstep + tid * 8;        // next k-tile source

    // precomputed swizzled LDS read offsets (bytes within a buffer)
    const int xs = (lane & 8) << 1;                      // bank-swizzle key
    const int ro0 = (lane * 32) ^ xs;
    const int ro1 = ((lane * 32) + 16) ^ xs;

    v8f acc[4] = {};
    int buf = 0;
    for (int k0 = 0; k0 < K; k0 += 32) {
        asm volatile("s_wait_asynccnt 0x0" ::: "memory");
        __syncthreads();
        if (k0 + 32 < K) {   // stage next k-tile into the other buffer
            unsigned dbase = buf ? ldsBase0 : ldsBase1;
            unsigned long long g = (unsigned long long)(size_t)bnext;
            asm volatile("global_load_async_to_lds_b128 %0, %1, off"
                         :: "v"(dbase + dstOff), "v"(g) : "memory");
            bnext += bstep;
        }

        FragAB a;
        a.h[0] = *(const v8bf*)(arow + k0 + sel);
        a.h[1] = *(const v8bf*)(arow + k0 + 8 + sel);
        __builtin_prefetch(arow + k0 + 32, 0, 0);   // global_prefetch_b8

        const char* bufBase = (const char*)&ldsB[buf][0];
        FragAB b0, b1, b2, b3;
        b0.h[0] = *(const v8bf*)(bufBase + 0 * 1024 + ro0);
        b0.h[1] = *(const v8bf*)(bufBase + 0 * 1024 + ro1);
        b1.h[0] = *(const v8bf*)(bufBase + 1 * 1024 + ro0);
        b1.h[1] = *(const v8bf*)(bufBase + 1 * 1024 + ro1);
        b2.h[0] = *(const v8bf*)(bufBase + 2 * 1024 + ro0);
        b2.h[1] = *(const v8bf*)(bufBase + 2 * 1024 + ro1);
        b3.h[0] = *(const v8bf*)(bufBase + 3 * 1024 + ro0);
        b3.h[1] = *(const v8bf*)(bufBase + 3 * 1024 + ro1);
        // scheduling fence: keep all 4 fragments live so the 8 ds_load_b128
        // issue together (one s_wait_dscnt) and the WMMAs run back-to-back
        asm volatile("" : "+v"(b0.v), "+v"(b1.v), "+v"(b2.v), "+v"(b3.v));

        acc[0] = __builtin_amdgcn_wmma_f32_16x16x32_bf16(
            false, a.v, false, b0.v, (short)0, acc[0], false, false);
        acc[1] = __builtin_amdgcn_wmma_f32_16x16x32_bf16(
            false, a.v, false, b1.v, (short)0, acc[1], false, false);
        acc[2] = __builtin_amdgcn_wmma_f32_16x16x32_bf16(
            false, a.v, false, b2.v, (short)0, acc[2], false, false);
        acc[3] = __builtin_amdgcn_wmma_f32_16x16x32_bf16(
            false, a.v, false, b3.v, (short)0, acc[3], false, false);
        buf ^= 1;
    }

    // C/D layout: VGPR r -> M = r (+8 for lanes 16-31), N = lane&15
    const int rowBase = (mTile << 4) + ((lane >> 4) << 3);
#pragma unroll
    for (int j = 0; j < 4; ++j) {
        const int col = ((ng << 2) + j) * 16 + (lane & 15);
        const float bb = bias ? bias[col] : 0.0f;
#pragma unroll
        for (int r = 0; r < 8; ++r)
            C[(size_t)(rowBase + r) * N + col] = acc[j][r] + bb;
    }
}

// ---------------------------------------------------------------------------
// BatchNorm statistics over rows (atomically accumulated sums / sumsqs)
// ---------------------------------------------------------------------------
__global__ __launch_bounds__(512)
void bn_stats_kernel(const float* __restrict__ H, float* __restrict__ bnSum,
                     float* __restrict__ bnSqs, int rowsPerBlock) {
    int col = threadIdx.x;
    int r0 = blockIdx.x * rowsPerBlock;
    float s = 0.0f, q = 0.0f;
    for (int r = r0; r < r0 + rowsPerBlock; ++r) {
        float v = H[(size_t)r * D_HID + col];
        s += v; q += v * v;
    }
    atomicAdd(&bnSum[col], s);
    atomicAdd(&bnSqs[col], q);
}

__global__ void bn_apply_kernel(const float* __restrict__ H, const float* __restrict__ bnSum,
                                const float* __restrict__ bnSqs, const float* __restrict__ gamma,
                                const float* __restrict__ beta, bf16_t* __restrict__ Hbf) {
    int idx = blockIdx.x * blockDim.x + threadIdx.x;
    if (idx >= B_ROWS * D_HID) return;
    int col = idx & (D_HID - 1);
    const float invM = 1.0f / (float)B_ROWS;
    float m   = bnSum[col] * invM;
    float var = bnSqs[col] * invM - m * m;
    float v = (H[idx] - m) * rsqrtf(var + BN_EPS) * gamma[col] + beta[col];
    Hbf[idx] = (bf16_t)fmaxf(v, 0.0f);
}

// ---------------------------------------------------------------------------
// In-place row L2 normalization (z_L = cld / sqrt(sum cld^2))
// ---------------------------------------------------------------------------
__global__ __launch_bounds__(256)
void rownorm_kernel(float* __restrict__ Z, int D) {
    int lane = threadIdx.x & 31;
    int gw = (blockIdx.x * blockDim.x + threadIdx.x) >> 5;
    int nw = (gridDim.x * blockDim.x) >> 5;
    for (int row = gw; row < B_ROWS; row += nw) {
        float* z = Z + (size_t)row * D;
        float s = 0.0f;
        for (int i = lane; i < D; i += 32) { float v = z[i]; s += v * v; }
#pragma unroll
        for (int m = 16; m >= 1; m >>= 1) s += __shfl_xor(s, m, 32);
        float inv = 1.0f / sqrtf(s);
        for (int i = lane; i < D; i += 32) z[i] *= inv;
    }
}

// ---------------------------------------------------------------------------
// Row cosine(z_H, y_other) accumulated into a scalar
// ---------------------------------------------------------------------------
__global__ __launch_bounds__(256)
void cosine_kernel(const float* __restrict__ ZH, const float* __restrict__ Yo,
                   float* __restrict__ accum) {
    int lane = threadIdx.x & 31;
    int gw = (blockIdx.x * blockDim.x + threadIdx.x) >> 5;
    int nw = (gridDim.x * blockDim.x) >> 5;
    for (int row = gw; row < B_ROWS; row += nw) {
        const float* a = ZH + (size_t)row * D_OUT;
        const float* b = Yo + (size_t)row * D_OUT;
        float d = 0.0f, na = 0.0f, nb = 0.0f;
        for (int i = lane; i < D_OUT; i += 32) {
            float av = a[i], bv = b[i];
            d += av * bv; na += av * av; nb += bv * bv;
        }
#pragma unroll
        for (int m = 16; m >= 1; m >>= 1) {
            d  += __shfl_xor(d, m, 32);
            na += __shfl_xor(na, m, 32);
            nb += __shfl_xor(nb, m, 32);
        }
        if (lane == 0) {
            float c = d / (fmaxf(sqrtf(na), 1e-8f) * fmaxf(sqrtf(nb), 1e-8f));
            atomicAdd(accum, c);
        }
    }
}

// ---------------------------------------------------------------------------
// K-means: centroid init (c = x[:10]), assign (argmin dist), segment-sum, update
// ---------------------------------------------------------------------------
__global__ void copyc_kernel(const float* __restrict__ X, float* __restrict__ Cc) {
    int i = blockIdx.x * blockDim.x + threadIdx.x;
    if (i < KCLUST * D_CLD) Cc[i] = X[i];
}

__global__ __launch_bounds__(256)
void kmeans_assign_kernel(const float* __restrict__ X, const float* __restrict__ Cc,
                          int* __restrict__ cl, float* __restrict__ cnt) {
    extern __shared__ float ldsA[];                 // 10*1024 centroids + 10 sumsq
    float* ldsC  = ldsA;
    float* ldsC2 = ldsA + KCLUST * D_CLD;
    for (int i = threadIdx.x; i < KCLUST * D_CLD; i += blockDim.x) ldsC[i] = Cc[i];
    __syncthreads();
    if (threadIdx.x < KCLUST) {
        float s = 0.0f;
        for (int j = 0; j < D_CLD; ++j) { float v = ldsC[threadIdx.x * D_CLD + j]; s += v * v; }
        ldsC2[threadIdx.x] = s;
    }
    __syncthreads();

    int lane = threadIdx.x & 31;
    int gw = (blockIdx.x * blockDim.x + threadIdx.x) >> 5;
    int nw = (gridDim.x * blockDim.x) >> 5;
    for (int row = gw; row < B_ROWS; row += nw) {
        const float* x = X + (size_t)row * D_CLD;
        float dot[KCLUST] = {};
        for (int i = lane; i < D_CLD; i += 32) {
            float xv = x[i];
#pragma unroll
            for (int k = 0; k < KCLUST; ++k) dot[k] += xv * ldsC[k * D_CLD + i];
        }
#pragma unroll
        for (int k = 0; k < KCLUST; ++k)
#pragma unroll
            for (int m = 16; m >= 1; m >>= 1) dot[k] += __shfl_xor(dot[k], m, 32);
        if (lane == 0) {
            // argmin of (x2 - 2 x.c + c2) == argmin of (c2 - 2 x.c)
            int best = 0; float bd = ldsC2[0] - 2.0f * dot[0];
#pragma unroll
            for (int k = 1; k < KCLUST; ++k) {
                float d = ldsC2[k] - 2.0f * dot[k];
                if (d < bd) { bd = d; best = k; }
            }
            cl[row] = best;
            atomicAdd(&cnt[best], 1.0f);
        }
    }
}

__global__ __launch_bounds__(256)
void kmeans_segsum_kernel(const float* __restrict__ X, const int* __restrict__ cl,
                          float* __restrict__ sums, int rowsPerBlock) {
    int col = blockIdx.x * blockDim.x + threadIdx.x;   // gridDim.x covers D_CLD
    int r0  = blockIdx.y * rowsPerBlock;
    float acc[KCLUST] = {};
    for (int r = r0; r < r0 + rowsPerBlock; ++r) {
        int k = cl[r];
        float xv = X[(size_t)r * D_CLD + col];
#pragma unroll
        for (int kk = 0; kk < KCLUST; ++kk) acc[kk] += (k == kk) ? xv : 0.0f;
    }
#pragma unroll
    for (int kk = 0; kk < KCLUST; ++kk)
        atomicAdd(&sums[kk * D_CLD + col], acc[kk]);
}

__global__ void kmeans_update_kernel(const float* __restrict__ sums, const float* __restrict__ cnt,
                                     float* __restrict__ Cc) {
    int i = blockIdx.x * blockDim.x + threadIdx.x;
    if (i >= KCLUST * D_CLD) return;
    int k = i / D_CLD;
    Cc[i] = sums[i] / fmaxf(cnt[k], 1.0f);
}

// ---------------------------------------------------------------------------
// Cross-entropy of aff = x @ c_other.T / T against labels (accumulated sum)
// ---------------------------------------------------------------------------
__global__ __launch_bounds__(256)
void ce_kernel(const float* __restrict__ X, const float* __restrict__ Cc,
               const int* __restrict__ labels, float* __restrict__ accum) {
    extern __shared__ float ldsE[];
    for (int i = threadIdx.x; i < KCLUST * D_CLD; i += blockDim.x) ldsE[i] = Cc[i];
    __syncthreads();
    int lane = threadIdx.x & 31;
    int gw = (blockIdx.x * blockDim.x + threadIdx.x) >> 5;
    int nw = (gridDim.x * blockDim.x) >> 5;
    const float invT = 1.0f / CLD_T;
    for (int row = gw; row < B_ROWS; row += nw) {
        const float* x = X + (size_t)row * D_CLD;
        float dot[KCLUST] = {};
        for (int i = lane; i < D_CLD; i += 32) {
            float xv = x[i];
#pragma unroll
            for (int k = 0; k < KCLUST; ++k) dot[k] += xv * ldsE[k * D_CLD + i];
        }
#pragma unroll
        for (int k = 0; k < KCLUST; ++k)
#pragma unroll
            for (int m = 16; m >= 1; m >>= 1) dot[k] += __shfl_xor(dot[k], m, 32);
        if (lane == 0) {
            float mx = -3.4e38f;
#pragma unroll
            for (int k = 0; k < KCLUST; ++k) { float l = dot[k] * invT; mx = fmaxf(mx, l); }
            float se = 0.0f;
#pragma unroll
            for (int k = 0; k < KCLUST; ++k) se += expf(dot[k] * invT - mx);
            float lse = mx + logf(se);
            int lab = labels[row];
            float ll = 0.0f;
#pragma unroll
            for (int k = 0; k < KCLUST; ++k) ll = (k == lab) ? dot[k] * invT : ll;
            atomicAdd(accum, lse - ll);
        }
    }
}

__global__ void finalize_kernel(const float* __restrict__ scal, const float* __restrict__ Lambda,
                                float* __restrict__ out) {
    if (threadIdx.x == 0 && blockIdx.x == 0) {
        const float invB = 1.0f / (float)B_ROWS;
        float lss  = 2.0f - scal[0] * invB - scal[1] * invB;
        float lcld = 0.5f * (scal[2] * invB + scal[3] * invB);
        out[0] = lss + Lambda[0] * lcld;
        out[1] = lss;
        out[2] = lcld;
    }
}

// ---------------------------------------------------------------------------
// Host launcher
// ---------------------------------------------------------------------------
extern "C" void kernel_launch(void* const* d_in, const int* in_sizes, int n_in,
                              void* d_out, int out_size, void* d_ws, size_t ws_size,
                              hipStream_t stream) {
    (void)in_sizes; (void)n_in; (void)out_size; (void)ws_size;
    const float* y1    = (const float*)d_in[0];
    const float* y2    = (const float*)d_in[1];
    const float* Lam   = (const float*)d_in[2];
    const float* W1    = (const float*)d_in[3];
    const float* b1    = (const float*)d_in[4];
    const float* gamma = (const float*)d_in[5];
    const float* beta  = (const float*)d_in[6];
    const float* W2    = (const float*)d_in[7];
    const float* b2    = (const float*)d_in[8];
    const float* Wc    = (const float*)d_in[9];
    float* out = (float*)d_out;

    char* p = (char*)d_ws;
    auto take = [&](size_t bytes) -> char* {
        char* r = p;
        p += (bytes + 255) & ~(size_t)255;
        return r;
    };
    bf16_t* bpW1  = (bf16_t*)take((size_t)D_IN  * D_HID * 2);
    bf16_t* bpW2  = (bf16_t*)take((size_t)D_HID * D_OUT * 2);
    bf16_t* bpWc  = (bf16_t*)take((size_t)D_IN  * D_CLD * 2);
    float*  cScale= (float*) take((size_t)D_CLD * 4);
    bf16_t* ybf   = (bf16_t*)take((size_t)B_ROWS * D_IN  * 2);
    bf16_t* ynbf  = (bf16_t*)take((size_t)B_ROWS * D_IN  * 2);
    float*  h     = (float*) take((size_t)B_ROWS * D_HID * 4);
    bf16_t* hbf   = (bf16_t*)take((size_t)B_ROWS * D_HID * 2);
    float*  z2H   = (float*) take((size_t)B_ROWS * D_OUT * 4);
    float*  z2L   = (float*) take((size_t)B_ROWS * D_CLD * 4);
    float*  c1    = (float*) take((size_t)KCLUST * D_CLD * 4);
    float*  c2    = (float*) take((size_t)KCLUST * D_CLD * 4);
    float*  sums  = (float*) take((size_t)KCLUST * D_CLD * 4);
    float*  cnt   = (float*) take(64);
    int*    cl1   = (int*)   take((size_t)B_ROWS * 4);
    int*    cl2   = (int*)   take((size_t)B_ROWS * 4);
    float*  bnSum = (float*) take((size_t)D_HID * 4);
    float*  bnSqs = (float*) take((size_t)D_HID * 4);
    float*  scal  = (float*) take(64);

    float* z1H = out + 3;
    float* z1L = out + 3 + (size_t)B_ROWS * D_OUT;

    // scalar accumulators: [0]=cos(z1H,y2) [1]=cos(z2H,y1) [2]=ce1 [3]=ce2
    zero_kernel<<<1, 64, 0, stream>>>(scal, 4);

    // Weight prep (once)
    colnorm_kernel<<<D_CLD / 256, 256, 0, stream>>>(Wc, cScale);
    {
        int f1 = (D_IN >> 5) * (D_HID >> 4);
        pack_b_kernel<<<(f1 * 32 + 255) / 256, 256, 0, stream>>>(W1, nullptr, bpW1, D_IN, D_HID);
        int f2 = (D_HID >> 5) * (D_OUT >> 4);
        pack_b_kernel<<<(f2 * 32 + 255) / 256, 256, 0, stream>>>(W2, nullptr, bpW2, D_HID, D_OUT);
        int fc = (D_IN >> 5) * (D_CLD >> 4);
        pack_b_kernel<<<(fc * 32 + 255) / 256, 256, 0, stream>>>(Wc, cScale, bpWc, D_IN, D_CLD);
    }

    // Two branches, sequentially (shared scratch)
    for (int br = 0; br < 2; ++br) {
        const float* y  = br ? y2 : y1;
        const float* yo = br ? y1 : y2;   // stop-grad target for cosine
        float* zH = br ? z2H : z1H;
        float* zL = br ? z2L : z1L;

        rowprep_kernel<<<256, 256, 0, stream>>>(y, ybf, ynbf);

        // h = y @ W1 + b1   (grid: (M/128) row groups x (N/64) col strips)
        gemm_bf16_wmma<<<(B_ROWS / 128) * (D_HID / 64), 256, 0, stream>>>(
            ybf, bpW1, b1, h, B_ROWS, D_HID, D_IN);

        zero_kernel<<<(D_HID + 255) / 256, 256, 0, stream>>>(bnSum, D_HID);
        zero_kernel<<<(D_HID + 255) / 256, 256, 0, stream>>>(bnSqs, D_HID);
        bn_stats_kernel<<<64, 512, 0, stream>>>(h, bnSum, bnSqs, B_ROWS / 64);
        bn_apply_kernel<<<(B_ROWS * D_HID + 255) / 256, 256, 0, stream>>>(
            h, bnSum, bnSqs, gamma, beta, hbf);

        // pred = relu_bn(h) @ W2 + b2
        gemm_bf16_wmma<<<(B_ROWS / 128) * (D_OUT / 64), 256, 0, stream>>>(
            hbf, bpW2, b2, zH, B_ROWS, D_OUT, D_HID);

        // cld = l2norm(y) @ l2norm(Wc); then row-normalize
        gemm_bf16_wmma<<<(B_ROWS / 128) * (D_CLD / 64), 256, 0, stream>>>(
            ynbf, bpWc, nullptr, zL, B_ROWS, D_CLD, D_IN);
        rownorm_kernel<<<256, 256, 0, stream>>>(zL, D_CLD);

        cosine_kernel<<<128, 256, 0, stream>>>(zH, yo, &scal[br]);
    }

    // K-means on z1_L and z2_L
    const size_t ldsAssign = (size_t)(KCLUST * D_CLD + 16) * 4;
    for (int br = 0; br < 2; ++br) {
        const float* x = br ? z2L : z1L;
        float* cc = br ? c2 : c1;
        int*   cl = br ? cl2 : cl1;
        copyc_kernel<<<(KCLUST * D_CLD + 255) / 256, 256, 0, stream>>>(x, cc);
        for (int it = 0; it < NITERS; ++it) {
            zero_kernel<<<(KCLUST * D_CLD + 255) / 256, 256, 0, stream>>>(sums, KCLUST * D_CLD);
            zero_kernel<<<1, 64, 0, stream>>>(cnt, KCLUST);
            kmeans_assign_kernel<<<64, 256, ldsAssign, stream>>>(x, cc, cl, cnt);
            kmeans_segsum_kernel<<<dim3(D_CLD / 256, 8), 256, 0, stream>>>(
                x, cl, sums, B_ROWS / 8);
            kmeans_update_kernel<<<(KCLUST * D_CLD + 255) / 256, 256, 0, stream>>>(sums, cnt, cc);
        }
    }

    // Cross-branch CLD cross-entropy
    const size_t ldsCE = (size_t)(KCLUST * D_CLD) * 4;
    ce_kernel<<<64, 256, ldsCE, stream>>>(z1L, c2, cl2, &scal[2]);
    ce_kernel<<<64, 256, ldsCE, stream>>>(z2L, c1, cl1, &scal[3]);

    finalize_kernel<<<1, 32, 0, stream>>>(scal, Lam, out);
}
